// MultiHeadAttention_40922448396371
// MI455X (gfx1250) — matrix operand order repaired and checked by reading, not compiled, not run on hardware
//
#include <hip/hip_runtime.h>
#include <hip/hip_bf16.h>

// Problem constants (from reference): B=4, T=2048, C=1024, H=16, D=64
#define BB 4
#define TT 2048
#define CC 1024
#define HH 16
#define DD 64

typedef __attribute__((ext_vector_type(16))) __bf16 bf16x16;
typedef __attribute__((ext_vector_type(8)))  __bf16 bf16x8;
typedef __attribute__((ext_vector_type(8)))  float  f32x8;

#if defined(__HIP_DEVICE_COMPILE__) && \
    __has_builtin(__builtin_amdgcn_global_load_async_to_lds_b128) && \
    __has_builtin(__builtin_amdgcn_s_wait_asynccnt)
#define USE_ASYNC_LDS 1
#else
#define USE_ASYNC_LDS 0
#endif

#define ASPC(n) __attribute__((address_space(n)))
// Builtin expects pointers to "int __vector(4)" (per hipcc diagnostic)
typedef int v4i_vs __attribute__((vector_size(4 * sizeof(int))));

__device__ __forceinline__ bf16x16 concat8(bf16x8 lo, bf16x8 hi) {
  return __builtin_shufflevector(lo, hi, 0, 1, 2, 3, 4, 5, 6, 7,
                                         8, 9, 10, 11, 12, 13, 14, 15);
}

// A-matrix 16x32 bf16: lane M = lane%16; element i ->
//   K = i + 16*half (contiguous, matching the B-fragment loads)
// From a row-major bf16 row pointer these are two contiguous 16B loads.
__device__ __forceinline__ bf16x16 loadA_row(const __bf16* rp, int half) {
  bf16x8 lo = *(const bf16x8*)(rp + 16 * half);
  bf16x8 hi = *(const bf16x8*)(rp + 16 * half + 8);
  return concat8(lo, hi);
}

// ============================================================================
// Conversion kernels (one-time, bandwidth-trivial: ~50MB @ 23.3 TB/s)
// ============================================================================
__global__ __launch_bounds__(256) void cvt_f32_bf16_kernel(
    const float* __restrict__ in, __bf16* __restrict__ out) {
  size_t base = ((size_t)blockIdx.x * 256 + threadIdx.x) * 8;
  float4 a = *(const float4*)(in + base);
  float4 b = *(const float4*)(in + base + 4);
  bf16x8 o;
  o[0] = (__bf16)a.x; o[1] = (__bf16)a.y; o[2] = (__bf16)a.z; o[3] = (__bf16)a.w;
  o[4] = (__bf16)b.x; o[5] = (__bf16)b.y; o[6] = (__bf16)b.z; o[7] = (__bf16)b.w;
  *(bf16x8*)(out + base) = o;
}

// wq/wk/wv [h][c][d] f32 -> Wt [mat][h][d][c] bf16 (pre-transposed so the
// B-fragment K-dim (c) is the contiguous axis)
__global__ __launch_bounds__(256) void cvt_wT_kernel(
    const float* __restrict__ wq, const float* __restrict__ wk,
    const float* __restrict__ wv, __bf16* __restrict__ Wt) {
  const int mat = blockIdx.z;
  const float* W = (mat == 0) ? wq : (mat == 1) ? wk : wv;
  __bf16* o = Wt + (size_t)mat * HH * CC * DD;
  int idx = blockIdx.x * 256 + threadIdx.x;      // over [h][d][c], c fastest
  int c = idx & (CC - 1);
  int d = (idx >> 10) & (DD - 1);
  int h = idx >> 16;
  o[idx] = (__bf16)W[((size_t)h * CC + c) * DD + d];
}

// ============================================================================
// Kernel 1: per-head QKV projection, LDS-free.
// Xb[8192,1024](bf16) x Wt[mat][h][d][c](bf16) -> Q/K bf16 [BH][T][D],
// V bf16 transposed [BH][D][T]. 8 waves/block, 16 rows x 64 cols per wave.
// ============================================================================
__global__ __launch_bounds__(256) void qkv_proj_kernel(
    const __bf16* __restrict__ Xb, const __bf16* __restrict__ Wt,
    __bf16* __restrict__ Qb, __bf16* __restrict__ Kb, __bf16* __restrict__ Vt)
{
  const int tid  = threadIdx.x;
  const int wid  = tid >> 5;
  const int lane = tid & 31;
  const int half = lane >> 4;
  const int lm   = lane & 15;
  const int rbase = blockIdx.x * 128;   // row in flattened [B*T]
  const int h     = blockIdx.y;
  const int mat   = blockIdx.z;         // 0=Q 1=K 2=V
  const __bf16* W = Wt + ((size_t)mat * HH + h) * CC * DD;   // [d][c]
  __bf16* outp    = (mat == 0) ? Qb : (mat == 1) ? Kb : Vt;

  const int row = rbase + wid * 16 + lm;
  const __bf16* xrow = Xb + (size_t)row * CC;

  f32x8 acc[4] = {{}, {}, {}, {}};
#pragma unroll 2
  for (int k0 = 0; k0 < CC; k0 += 32) {
    bf16x16 a = loadA_row(xrow + k0, half);
#pragma unroll
    for (int nt = 0; nt < 4; ++nt) {
      bf16x16 b = *(const bf16x16*)(W + (size_t)(nt * 16 + lm) * CC + k0 + 16 * half);
      acc[nt] = __builtin_amdgcn_wmma_f32_16x16x32_bf16(
          false, a, false, b, (short)0, acc[nt], false, false);
    }
  }
#pragma unroll
  for (int nt = 0; nt < 4; ++nt) {
    int d = nt * 16 + lm;
#pragma unroll
    for (int i = 0; i < 8; ++i) {
      int R  = rbase + wid * 16 + i + 8 * half;
      int b_ = R >> 11;
      int t  = R & (TT - 1);
      int bh = b_ * HH + h;
      __bf16 v = (__bf16)acc[nt][i];
      if (mat != 2) outp[((size_t)bh * TT + t) * DD + d] = v;   // Q,K row-major
      else          outp[((size_t)bh * DD + d) * TT + t] = v;   // V transposed
    }
  }
}

// ============================================================================
// Async staging of one 32-key K block (contiguous 4KB) and V^T block
// (64 rows x 64B) into LDS: 16x global_load_async_to_lds_b128 per wave.
// ============================================================================
#if USE_ASYNC_LDS
__device__ __forceinline__ void stage_kv(const __bf16* kg, const __bf16* vg,
                                         __bf16* lk, __bf16* lv, int lane) {
#pragma unroll
  for (int i = 0; i < 8; ++i) {   // K: 8 x 512B = 4KB contiguous
    __builtin_amdgcn_global_load_async_to_lds_b128(
        (ASPC(1) v4i_vs*)(kg + i * 256 + lane * 8),
        (ASPC(3) v4i_vs*)(lk + i * 256 + lane * 8), 0, 0);
  }
#pragma unroll
  for (int i = 0; i < 8; ++i) {   // V^T: 8 d-rows per instr, 64B per row
    const int d = i * 8 + (lane >> 2);
    const int toff = (lane & 3) * 8;
    __builtin_amdgcn_global_load_async_to_lds_b128(
        (ASPC(1) v4i_vs*)(vg + (size_t)d * TT + toff),
        (ASPC(3) v4i_vs*)(lv + d * 32 + toff), 0, 0);
  }
}
#endif

// ============================================================================
// Kernel 2: causal flash attention, one wave (32 threads) per 16-query tile.
// Key blocks of 32 -> two 16x16 score tiles; online softmax; P transposed
// through LDS to WMMA A layout; 4 PV WMMAs accumulate O[16x64].
// K/V blocks are double-buffered into LDS with the async-DMA engine
// (ASYNCcnt) so copy latency overlaps the WMMA + softmax work.
// ============================================================================
__global__ __launch_bounds__(32) void flash_attn_kernel(
    const __bf16* __restrict__ Qb, const __bf16* __restrict__ Kb,
    const __bf16* __restrict__ Vt, __bf16* __restrict__ AO)
{
  __shared__ __bf16 Plds[16 * 32];
#if USE_ASYNC_LDS
  __shared__ __align__(32) __bf16 ldsK[2][32 * 64];
  __shared__ __align__(32) __bf16 ldsV[2][64 * 32];
#endif
  const int lane = threadIdx.x;
  const int half = lane >> 4;
  const int lm   = lane & 15;
  const int qt   = blockIdx.x;           // query tile (16 rows)
  const int bh   = blockIdx.y;           // b*H + h
  const int b_   = bh >> 4;
  const int h    = bh & 15;
  const float scale = 0.03125f;          // C^-0.5 = 1024^-0.5 (per reference)

  const __bf16* Kgb = Kb + (size_t)bh * TT * DD;   // key rows base
  const __bf16* Vgb = Vt + (size_t)bh * DD * TT;   // V^T base

  // Q A-fragments: row = qt*16 + lm; two 32-wide K chunks over D=64.
  const __bf16* qrow = Qb + ((size_t)bh * TT + qt * 16 + lm) * DD;
  bf16x16 qa[2];
#pragma unroll
  for (int c = 0; c < 2; ++c) qa[c] = loadA_row(qrow + c * 32, half);

  float m[8], ls[8];
#pragma unroll
  for (int i = 0; i < 8; ++i) { m[i] = -3.0e38f; ls[i] = 0.0f; }
  f32x8 acc[4] = {{}, {}, {}, {}};

  const int nkb = (qt >> 1) + 1;          // 32-key blocks under causal mask

#if USE_ASYNC_LDS
  stage_kv(Kgb, Vgb, ldsK[0], ldsV[0], lane);   // preload block 0
#endif

  for (int j = 0; j < nkb; ++j) {
    const int j0 = j * 32;
#if USE_ASYNC_LDS
    const int cur = j & 1;
    if (j + 1 < nkb) {
      // kick off next block into the other buffer, then wait for current only
      stage_kv(Kgb + (size_t)(j0 + 32) * DD, Vgb + j0 + 32,
               ldsK[cur ^ 1], ldsV[cur ^ 1], lane);
      __builtin_amdgcn_s_wait_asynccnt(16);
    } else {
      __builtin_amdgcn_s_wait_asynccnt(0);
    }
    __syncthreads();                       // order DMA'd LDS data vs ds reads
    const __bf16* kbase = ldsK[cur];       // [key][64]
    const __bf16* vbase = ldsV[cur];       // [d][32]
    const int vstr = 32;
#else
    if (j + 1 < nkb) {
      __builtin_prefetch(Kgb + (size_t)(j0 + 32 + lane) * DD, 0, 3);
      __builtin_prefetch(Vgb + (size_t)(lane * 2) * TT + j0 + 32, 0, 3);
    }
    const __bf16* kbase = Kgb + (size_t)j0 * DD;   // [key][64]
    const __bf16* vbase = Vgb + j0;                // [d][TT]
    const int vstr = TT;
#endif

    f32x8 S0 = {}, S1 = {};
#pragma unroll
    for (int c = 0; c < 2; ++c) {
      int d0 = c * 32;
      bf16x16 kb0 = *(const bf16x16*)(kbase + (size_t)lm * DD + d0 + 16 * half);
      bf16x16 kb1 = *(const bf16x16*)(kbase + (size_t)(16 + lm) * DD + d0 + 16 * half);
      S0 = __builtin_amdgcn_wmma_f32_16x16x32_bf16(false, qa[c], false, kb0, (short)0, S0, false, false);
      S1 = __builtin_amdgcn_wmma_f32_16x16x32_bf16(false, qa[c], false, kb1, (short)0, S1, false, false);
    }
    const int key0 = j0 + lm, key1 = j0 + 16 + lm;
#pragma unroll
    for (int i = 0; i < 8; ++i) {
      const int t = qt * 16 + i + 8 * half;          // query row for element i
      float s0 = (key0 <= t) ? S0[i] * scale : -1.0e30f;
      float s1 = (key1 <= t) ? S1[i] * scale : -1.0e30f;
      float v = fmaxf(s0, s1);
#pragma unroll
      for (int off = 1; off < 16; off <<= 1) v = fmaxf(v, __shfl_xor(v, off, 16));
      const float mn = fmaxf(m[i], v);
      const float p0 = __expf(s0 - mn);
      const float p1 = __expf(s1 - mn);
      float ps = p0 + p1;
#pragma unroll
      for (int off = 1; off < 16; off <<= 1) ps += __shfl_xor(ps, off, 16);
      const float alpha = __expf(m[i] - mn);
      ls[i] = ls[i] * alpha + ps;
      m[i]  = mn;
#pragma unroll
      for (int nt = 0; nt < 4; ++nt) acc[nt][i] *= alpha;
      Plds[(i + 8 * half) * 32 + lm]      = (__bf16)p0;
      Plds[(i + 8 * half) * 32 + 16 + lm] = (__bf16)p1;
    }
    __syncthreads();   // single-wave block: barrier -> s_nop; orders LDS
    bf16x16 pa = loadA_row(Plds + lm * 32, half);
#pragma unroll
    for (int nt = 0; nt < 4; ++nt) {
      int d = nt * 16 + lm;
      bf16x16 vb = *(const bf16x16*)(vbase + (size_t)d * vstr + 16 * half);
      acc[nt] = __builtin_amdgcn_wmma_f32_16x16x32_bf16(false, pa, false, vb, (short)0, acc[nt], false, false);
    }
    __syncthreads();
  }
  // normalize and write AO[b][t][h*D + d] (bf16, head-concat layout)
#pragma unroll
  for (int i = 0; i < 8; ++i) {
    const int t = qt * 16 + i + 8 * half;
    const float inv = 1.0f / ls[i];
#pragma unroll
    for (int nt = 0; nt < 4; ++nt) {
      int d = nt * 16 + lm;
      AO[((size_t)b_ * TT + t) * CC + h * DD + d] = (__bf16)(acc[nt][i] * inv);
    }
  }
}

// ============================================================================
// Kernel 3: output projection  out = AO @ w_proj^T + b_proj  (f32 out).
// LDS-free: bf16 w_proj rows are already contiguous in the B-fragment K-dim.
// ============================================================================
__global__ __launch_bounds__(256) void out_proj_kernel(
    const __bf16* __restrict__ AO, const __bf16* __restrict__ WPb,
    const float* __restrict__ bp, float* __restrict__ out)
{
  const int tid  = threadIdx.x;
  const int wid  = tid >> 5;
  const int lane = tid & 31;
  const int half = lane >> 4;
  const int lm   = lane & 15;
  const int rbase = blockIdx.x * 128;
  const int nbase = blockIdx.y * 64;

  const int row = rbase + wid * 16 + lm;
  const __bf16* arow = AO + (size_t)row * CC;

  f32x8 acc[4] = {{}, {}, {}, {}};
#pragma unroll 2
  for (int c0 = 0; c0 < CC; c0 += 32) {
    bf16x16 a = loadA_row(arow + c0, half);
#pragma unroll
    for (int nt = 0; nt < 4; ++nt) {
      int n = nbase + nt * 16 + lm;
      bf16x16 b = *(const bf16x16*)(WPb + (size_t)n * CC + c0 + 16 * half);
      acc[nt] = __builtin_amdgcn_wmma_f32_16x16x32_bf16(
          false, a, false, b, (short)0, acc[nt], false, false);
    }
  }
#pragma unroll
  for (int nt = 0; nt < 4; ++nt) {
    int n = nbase + nt * 16 + lm;
    float bias = bp[n];
#pragma unroll
    for (int i = 0; i < 8; ++i) {
      int r = rbase + wid * 16 + i + 8 * half;
      out[(size_t)r * CC + n] = acc[nt][i] + bias;
    }
  }
}

extern "C" void kernel_launch(void* const* d_in, const int* in_sizes, int n_in,
                              void* d_out, int out_size, void* d_ws, size_t ws_size,
                              hipStream_t stream) {
  const float* x  = (const float*)d_in[0];
  const float* wq = (const float*)d_in[1];
  const float* wk = (const float*)d_in[2];
  const float* wv = (const float*)d_in[3];
  const float* wp = (const float*)d_in[4];
  const float* bp = (const float*)d_in[5];
  float* out = (float*)d_out;

  // Workspace (bf16): Xb, Wt(3 QKV transposed), WPb, Q, K, V^T, AO  (~92 MB)
  const size_t N  = (size_t)BB * HH * TT * DD;   // 8,388,608 (= B*T*C)
  const size_t NW = (size_t)HH * CC * DD;        // 1,048,576 per QKV matrix
  __bf16* Xb  = (__bf16*)d_ws;
  __bf16* Wt  = Xb + N;
  __bf16* WPb = Wt + 3 * NW;
  __bf16* Qb  = WPb + (size_t)CC * CC;
  __bf16* Kb  = Qb + N;
  __bf16* Vt  = Kb + N;
  __bf16* AO  = Vt + N;

  // 0) one-time conversions
  cvt_f32_bf16_kernel<<<dim3((unsigned)(N / (8 * 256))), 256, 0, stream>>>(x, Xb);
  cvt_wT_kernel<<<dim3((unsigned)(NW / 256), 1, 3), 256, 0, stream>>>(wq, wk, wv, Wt);
  cvt_f32_bf16_kernel<<<dim3((unsigned)((size_t)CC * CC / (8 * 256))), 256, 0, stream>>>(wp, WPb);

  // 1) QKV projections: grid = (8192/128 row-tiles, 16 heads, 3 matrices)
  qkv_proj_kernel<<<dim3(64, 16, 3), 256, 0, stream>>>(Xb, Wt, Qb, Kb, Vt);
  // 2) Flash attention: grid = (T/16 q-tiles, B*H), one wave per block
  flash_attn_kernel<<<dim3(TT / 16, BB * HH), 32, 0, stream>>>(Qb, Kb, Vt, AO);
  // 3) Output projection: grid = (8192/128, 1024/64)
  out_proj_kernel<<<dim3(64, 16), 256, 0, stream>>>(AO, WPb, bp, out);
}
